// SAGE_76673756168338
// MI455X (gfx1250) — matrix operand order repaired and checked by reading, not compiled, not run on hardware
//
#include <hip/hip_runtime.h>

#define N0 1000000
#define N1 200000
#define N2 20000
#define N3 2048
#define IN 128
#define HID 256
#define OUT 47
#define OUTP 48   // padded output cols for layer 2 (pad col zeroed in ws)

typedef __attribute__((ext_vector_type(16))) _Float16 v16h;
typedef __attribute__((ext_vector_type(8)))  float    v8f;

union AF { v16h v; _Float16 h[16]; };
union BF { v16h v; _Float16 h[16]; uint4 q[2]; };
union CF { v8f  v; float f[8]; };

// ---------------------------------------------------------------------------
// Convert f32 weight (K x N row major) -> f16 transposed (N x K) so a WMMA B
// fragment (16 consecutive K values at fixed N) is one contiguous 32B read.
// ---------------------------------------------------------------------------
__global__ void wt_convert(const float* __restrict__ W, _Float16* __restrict__ WT,
                           int K, int N) {
    int idx = blockIdx.x * blockDim.x + threadIdx.x;
    if (idx < K * N) {
        int k = idx / N;
        int n = idx % N;
        WT[(size_t)n * K + k] = (_Float16)W[idx];
    }
}

// ---------------------------------------------------------------------------
// Edge scatter: one wave per edge, coalesced row gather, f32 atomics into the
// aggregation buffer (102 MB for layer 0 -> resolves in the 192 MB L2).
// ---------------------------------------------------------------------------
__global__ __launch_bounds__(256)
void scatter_mean_acc(const float* __restrict__ H, const int* __restrict__ src,
                      const int* __restrict__ dst, float* __restrict__ Agg,
                      float* __restrict__ Deg, int E, int F) {
    int wave = blockIdx.x * (blockDim.x >> 5) + (threadIdx.x >> 5);
    int lane = threadIdx.x & 31;
    if (wave >= E) return;
    int s = src[wave];
    int d = dst[wave];
    const float* hs = H + (size_t)s * F;
    float* ag = Agg + (size_t)d * F;
    for (int f = lane; f < F; f += 32)
        atomicAdd(&ag[f], hs[f]);
    if (lane == 0)
        atomicAdd(&Deg[d], 1.0f);
}

// ---------------------------------------------------------------------------
// Fused SAGE layer GEMM (block = 8 waves sharing one 16*NT column chunk):
//   Out[m][n] = act( H[m,:]@Ws[:,n] + (Agg[m,:]/max(Deg[m],1))@Wn[:,n] + b[n] )
// Per k-step the block stages the 32-wide K slice of both (transposed f16)
// weight matrices into LDS once (40-half row stride -> conflict-free
// ds_load_b128 fragment reads). All 2*NT B fragments are then batch-loaded
// into registers as one ds_load clause so the 2*NT v_wmma_f32_16x16x32_f16
// drain with staggered dscnt waits instead of per-op full-latency stalls.
// Degree normalization is fused into the A-neighbor fragment; bias+ReLU into
// the C/D epilogue. K/NT compile time; padded weight cols -> unguarded loads.
// ---------------------------------------------------------------------------
template <int NT, int K>
__global__ __launch_bounds__(256)
void sage_gemm(const float* __restrict__ H, const float* __restrict__ Agg,
               const float* __restrict__ Deg,
               const _Float16* __restrict__ WsT, const _Float16* __restrict__ WnT,
               const float* __restrict__ bias, float* __restrict__ Out,
               int M, int Nout, int doRelu) {
    constexpr int LSTR = 40;                       // padded LDS row stride (halves)
    __shared__ __align__(16) _Float16 sB[2][NT * 16][LSTR];

    const int tid  = threadIdx.x;
    const int lane = tid & 31;
    const int half = lane >> 4;                    // 0: lanes 0-15, 1: lanes 16-31
    const int r    = lane & 15;

    const int m0      = (blockIdx.x * 8 + (tid >> 5)) * 16;
    const bool active = (m0 < M);
    const int mA      = min(m0 + r, M - 1);        // clamped: inactive waves still
                                                   // load valid memory + hit barriers
    const int n0      = blockIdx.y * (16 * NT);

    const float invdeg = __builtin_amdgcn_rcpf(fmaxf(Deg[mA], 1.0f));
    const float* hrow  = H   + (size_t)mA * K;
    const float* arow  = Agg + (size_t)mA * K;

    // staging assignment: 2 threads per weight column, 16 halves (32B) each
    const int sc = tid >> 1;                        // column within chunk
    const int sp = tid & 1;                         // which 16-half part

    CF acc[NT];
#pragma unroll
    for (int t = 0; t < NT; ++t)
#pragma unroll
        for (int e = 0; e < 8; ++e) acc[t].f[e] = 0.0f;

    for (int ks = 0; ks < K / 32; ++ks) {
        const int k0 = ks * 32;

        // ---- stage B slice (both matrices) into LDS ----
        if (sc < NT * 16) {
            const uint4* ps = (const uint4*)(WsT + (size_t)(n0 + sc) * K + k0 + sp * 16);
            const uint4* pn = (const uint4*)(WnT + (size_t)(n0 + sc) * K + k0 + sp * 16);
            *(uint4*)&sB[0][sc][sp * 16] = ps[0];
            *(uint4*)&sB[1][sc][sp * 16] = pn[0];
        }
        __syncthreads();

        // ---- A fragments (16x32 f16 layout), f32->f16 with fused mean ----
        AF as, an;
        const int base0 = k0 + half * 16;
        const int base1 = k0 + half * 16 + 8;
#pragma unroll
        for (int j = 0; j < 8; ++j) {
            as.h[j]     = (_Float16)hrow[base0 + j];
            as.h[j + 8] = (_Float16)hrow[base1 + j];
            an.h[j]     = (_Float16)(arow[base0 + j] * invdeg);
            an.h[j + 8] = (_Float16)(arow[base1 + j] * invdeg);
        }

        // ---- batch-load all 2*NT B fragments (one ds_load clause) ----
        BF bs[NT], bn[NT];
#pragma unroll
        for (int t = 0; t < NT; ++t) {
            const uint4* ls = (const uint4*)&sB[0][t * 16 + r][half * 16];
            bs[t].q[0] = ls[0]; bs[t].q[1] = ls[1];
        }
#pragma unroll
        for (int t = 0; t < NT; ++t) {
            const uint4* ln = (const uint4*)&sB[1][t * 16 + r][half * 16];
            bn[t].q[0] = ln[0]; bn[t].q[1] = ln[1];
        }

        // ---- 2*NT WMMAs, consumed in load order (monotone dscnt waits) ----
#pragma unroll
        for (int t = 0; t < NT; ++t)
            acc[t].v = __builtin_amdgcn_wmma_f32_16x16x32_f16(
                false, as.v, false, bs[t].v, (short)0, acc[t].v, false, false);
#pragma unroll
        for (int t = 0; t < NT; ++t)
            acc[t].v = __builtin_amdgcn_wmma_f32_16x16x32_f16(
                false, an.v, false, bn[t].v, (short)0, acc[t].v, false, false);

        __syncthreads();
    }

    if (!active) return;

    // ---- epilogue: C/D layout (elem e -> row m0+e+8*half, col n0+t*16+r) ----
#pragma unroll
    for (int t = 0; t < NT; ++t) {
        int n = n0 + t * 16 + r;
        if (n >= Nout) continue;
        float bv = bias[n];
#pragma unroll
        for (int e = 0; e < 8; ++e) {
            int row = m0 + e + half * 8;
            float val = acc[t].f[e] + bv;
            if (doRelu) val = fmaxf(val, 0.0f);
            Out[(size_t)row * Nout + n] = val;
        }
    }
}

extern "C" void kernel_launch(void* const* d_in, const int* in_sizes, int n_in,
                              void* d_out, int out_size, void* d_ws, size_t ws_size,
                              hipStream_t stream) {
    const float* x    = (const float*)d_in[0];
    const int*   src0 = (const int*)d_in[1];
    const int*   dst0 = (const int*)d_in[2];
    const int*   src1 = (const int*)d_in[3];
    const int*   dst1 = (const int*)d_in[4];
    const int*   src2 = (const int*)d_in[5];
    const int*   dst2 = (const int*)d_in[6];
    const float* Ws0  = (const float*)d_in[7];
    const float* Wn0  = (const float*)d_in[8];
    const float* b0   = (const float*)d_in[9];
    const float* Ws1  = (const float*)d_in[10];
    const float* Wn1  = (const float*)d_in[11];
    const float* b1   = (const float*)d_in[12];
    const float* Ws2  = (const float*)d_in[13];
    const float* Wn2  = (const float*)d_in[14];
    const float* b2   = (const float*)d_in[15];

    const int E0 = in_sizes[1];
    const int E1 = in_sizes[3];
    const int E2 = in_sizes[5];

    // --- workspace bump allocator (256B aligned regions) ---
    char* p = (char*)d_ws;
    auto alloc = [&](size_t bytes) -> char* {
        char* r = p;
        p += (bytes + 255) & ~(size_t)255;
        return r;
    };
    // zeroed region: agg/deg accumulators + padded layer-2 weights
    float*    agg0 = (float*)alloc((size_t)N1 * IN  * 4);
    float*    deg0 = (float*)alloc((size_t)N1 * 4);
    float*    agg1 = (float*)alloc((size_t)N2 * HID * 4);
    float*    deg1 = (float*)alloc((size_t)N2 * 4);
    float*    agg2 = (float*)alloc((size_t)N3 * HID * 4);
    float*    deg2 = (float*)alloc((size_t)N3 * 4);
    _Float16* WsT2 = (_Float16*)alloc((size_t)OUTP * HID * 2);  // pad col stays 0
    _Float16* WnT2 = (_Float16*)alloc((size_t)OUTP * HID * 2);
    size_t zero_bytes = (size_t)(p - (char*)d_ws);
    // activations + f16 transposed weights (layers 0/1)
    float*    h1   = (float*)alloc((size_t)N1 * HID * 4);
    float*    h2   = (float*)alloc((size_t)N2 * HID * 4);
    _Float16* WsT0 = (_Float16*)alloc((size_t)HID * IN  * 2);
    _Float16* WnT0 = (_Float16*)alloc((size_t)HID * IN  * 2);
    _Float16* WsT1 = (_Float16*)alloc((size_t)HID * HID * 2);
    _Float16* WnT1 = (_Float16*)alloc((size_t)HID * HID * 2);

    hipMemsetAsync(d_ws, 0, zero_bytes, stream);

    // --- weight convert + transpose (tiny, L2 resident afterwards) ---
    {
        int n01 = IN * HID, n11 = HID * HID, n21 = HID * OUT;
        wt_convert<<<(n01 + 255) / 256, 256, 0, stream>>>(Ws0, WsT0, IN, HID);
        wt_convert<<<(n01 + 255) / 256, 256, 0, stream>>>(Wn0, WnT0, IN, HID);
        wt_convert<<<(n11 + 255) / 256, 256, 0, stream>>>(Ws1, WsT1, HID, HID);
        wt_convert<<<(n11 + 255) / 256, 256, 0, stream>>>(Wn1, WnT1, HID, HID);
        wt_convert<<<(n21 + 255) / 256, 256, 0, stream>>>(Ws2, WsT2, HID, OUT);
        wt_convert<<<(n21 + 255) / 256, 256, 0, stream>>>(Wn2, WnT2, HID, OUT);
    }

    // --- layer 0: x(1M x 128) -> h1(200K x 256) ---
    scatter_mean_acc<<<(E0 + 7) / 8, 256, 0, stream>>>(x, src0, dst0, agg0, deg0, E0, IN);
    {
        dim3 grid((N1 / 16 + 7) / 8, HID / 128);
        sage_gemm<8, IN><<<grid, 256, 0, stream>>>(x, agg0, deg0, WsT0, WnT0, b0, h1,
                                                   N1, HID, 1);
    }
    // --- layer 1: h1 -> h2(20K x 256) ---
    scatter_mean_acc<<<(E1 + 7) / 8, 256, 0, stream>>>(h1, src1, dst1, agg1, deg1, E1, HID);
    {
        dim3 grid((N2 / 16 + 7) / 8, HID / 128);
        sage_gemm<8, HID><<<grid, 256, 0, stream>>>(h1, agg1, deg1, WsT1, WnT1, b1, h2,
                                                    N2, HID, 1);
    }
    // --- layer 2: h2 -> out(2048 x 47), no relu, padded weight cols ---
    scatter_mean_acc<<<(E2 + 7) / 8, 256, 0, stream>>>(h2, src2, dst2, agg2, deg2, E2, HID);
    {
        dim3 grid((N3 / 16 + 7) / 8, 1);
        sage_gemm<3, HID><<<grid, 256, 0, stream>>>(h2, agg2, deg2, WsT2, WnT2, b2,
                                                    (float*)d_out, N3, OUT, 0);
    }
}
